// DualLaplacianBlock_15075335209398
// MI455X (gfx1250) — matrix operand order repaired and checked by reading, not compiled, not run on hardware
//
#include <hip/hip_runtime.h>
#include <hip/hip_bf16.h>

// ---------------------------------------------------------------------------
// DualLaplacianBlock on MI455X (gfx1250, wave32, WMMA bf16 16x16x32).
// All heavy GEMMs (projections, Gram matrices, Laplacian applies, output
// projection) run through v_wmma_f32_16x16x32_bf16 with f32 accumulation.
// The n x n adjacencies are never materialized in HBM: they are recomputed
// (degree pass) and strip-fused through LDS (apply pass), keeping the whole
// intermediate working set (~50 MB) resident in the 192 MB L2.
// Workspace requirement: 6 * 8.39 MB + 128 KB ~= 50.5 MB.
// ---------------------------------------------------------------------------

#define DEVINL __device__ __forceinline__

typedef __attribute__((ext_vector_type(16))) __bf16 v16bf;
typedef __attribute__((ext_vector_type(8)))  __bf16 v8bf;
typedef __attribute__((ext_vector_type(8)))  float  v8f;

constexpr int   BATCH   = 4;
constexpr int   NN      = 2048;   // tokens per batch
constexpr int   DD      = 512;    // feature dim
constexpr float EPSCOS  = 1e-8f;
constexpr float EPSDEG  = 1e-6f;
constexpr float TWAKE   = 0.1f;

DEVINL v8f wmma_bf16(v16bf a, v16bf b, v8f c) {
    return __builtin_amdgcn_wmma_f32_16x16x32_bf16(
        /*neg_a=*/false, a, /*neg_b=*/false, b,
        /*c_mod=*/(short)0, c, /*reuse_a=*/false, /*reuse_b=*/false);
}

// A-operand fragment (16x32 bf16, row-major source, stride ld elements).
// Layout (ISA 7.12.2): lanes 0-15 hold row M=lane, K = [0..7] and [16..23];
// lanes 16-31 hold K = [8..15] and [24..31].
DEVINL v16bf load_a_bf16(const __bf16* base, int ld, int row0, int k0, int lane) {
    int row = row0 + (lane & 15);
    int klo = k0 + ((lane & 16) ? 8 : 0);
    const __bf16* p = base + (size_t)row * ld + klo;
    v8bf lo = *(const v8bf*)(p);
    v8bf hi = *(const v8bf*)(p + 16);
    return __builtin_shufflevector(lo, hi, 0,1,2,3,4,5,6,7,8,9,10,11,12,13,14,15);
}

// Same pattern but converting from f32 source (h and W are f32 in HBM).
DEVINL v16bf load_a_f32(const float* base, int ld, int row0, int k0, int lane) {
    int row = row0 + (lane & 15);
    int klo = k0 + ((lane & 16) ? 8 : 0);
    const float* p = base + (size_t)row * ld + klo;
    v16bf r;
#pragma unroll
    for (int t = 0; t < 8; ++t) { r[t] = (__bf16)p[t]; r[t + 8] = (__bf16)p[t + 16]; }
    return r;
}

// B-operand fragment (32x16 bf16). Source matrix stores each B-column as a
// row of length-K contiguous elements (i.e. Bm[col][k], stride ld).
// Layout: lanes 0-15 hold K=[0..7],[16..23] of column lane; lanes 16-31 hold K=[8..15],[24..31].
DEVINL v16bf load_b_bf16(const __bf16* base, int ld, int k0, int col0, int lane) {
    int col = col0 + (lane & 15);
    int kb  = k0 + ((lane & 16) ? 8 : 0);
    const __bf16* p = base + (size_t)col * ld + kb;
    v8bf lo = *(const v8bf*)(p);
    v8bf hi = *(const v8bf*)(p + 16);
    return __builtin_shufflevector(lo, hi, 0,1,2,3,4,5,6,7,8,9,10,11,12,13,14,15);
}

DEVINL v16bf load_b_f32(const float* base, int ld, int k0, int col0, int lane) {
    int col = col0 + (lane & 15);
    int kb  = k0 + ((lane & 16) ? 8 : 0);
    const float* p = base + (size_t)col * ld + kb;
    v16bf r;
#pragma unroll
    for (int t = 0; t < 8; ++t) { r[t] = (__bf16)p[t]; r[t + 8] = (__bf16)p[t + 16]; }
    return r;
}

// ---------------------------------------------------------------------------
// Kernel 1: z = h @ W^T  (torch-style W[out,in]). Writes bf16 z row-major
// [B*N, D] and bf16 zT [B, D, N] for later contiguous-K B-operand loads.
// 8 waves/block, one 16x16 tile per wave. grid = 16384/8 = 2048.
// ---------------------------------------------------------------------------
__global__ __launch_bounds__(256) void proj_kernel(
    const float* __restrict__ h, const float* __restrict__ W,
    __bf16* __restrict__ z, __bf16* __restrict__ zT) {
    int wid  = threadIdx.x >> 5;
    int lane = threadIdx.x & 31;
    int tile = blockIdx.x * 8 + wid;          // 0 .. 16383
    int m0   = (tile >> 5) * 16;              // row tile over B*N
    int e0   = (tile & 31) * 16;              // col tile over D

    v8f c = {};
#pragma unroll 4
    for (int k0 = 0; k0 < DD; k0 += 32)
        c = wmma_bf16(load_a_f32(h, DD, m0, k0, lane),
                      load_b_f32(W, DD, k0, e0, lane), c);

    int rbase = m0 + ((lane & 16) ? 8 : 0);
    int col   = e0 + (lane & 15);
#pragma unroll
    for (int r = 0; r < 8; ++r) {
        int   row = rbase + r;
        __bf16 v  = (__bf16)c[r];
        z[(size_t)row * DD + col] = v;
        int b = row >> 11;                    // row / N
        int i = row & (NN - 1);               // row % N
        zT[((size_t)b * DD + col) * NN + i] = v;
    }
}

// ---------------------------------------------------------------------------
// Kernel 2: per-row stats. invn_l[r] = 1/max(||z_l[r]||, eps),
// sq_g[r] = ||z_g[r]||^2. One thread per row.
// ---------------------------------------------------------------------------
__global__ __launch_bounds__(256) void stats_kernel(
    const __bf16* __restrict__ z_l, const __bf16* __restrict__ z_g,
    float* __restrict__ invn_l, float* __restrict__ sq_g) {
    int row = blockIdx.x * 256 + threadIdx.x;
    if (row >= BATCH * NN) return;
    const __bf16* pl = z_l + (size_t)row * DD;
    const __bf16* pg = z_g + (size_t)row * DD;
    float sl = 0.f, sg = 0.f;
#pragma unroll 8
    for (int k = 0; k < DD; ++k) {
        float a = (float)pl[k]; sl += a * a;
        float b = (float)pg[k]; sg += b * b;
    }
    invn_l[row] = 1.0f / fmaxf(sqrtf(sl), EPSCOS);
    sq_g[row]   = sg;
}

// ---------------------------------------------------------------------------
// Kernel 3: degrees of both adjacencies via on-the-fly Gram tiles.
// One wave per (b, 16-row i-tile); loop j-tiles; 2 WMMA K-chains per tile;
// elementwise transform in f32 on the C-layout; shfl_xor row-sum reduction.
// Writes invs = rsqrt(max(deg, eps)). grid = 512/8 = 64.
// ---------------------------------------------------------------------------
__global__ __launch_bounds__(256) void degree_kernel(
    const __bf16* __restrict__ z_l, const __bf16* __restrict__ z_g,
    const float* __restrict__ invn_l, const float* __restrict__ sq_g,
    float* __restrict__ invs_l, float* __restrict__ invs_g) {
    int wid  = threadIdx.x >> 5;
    int lane = threadIdx.x & 31;
    int it   = blockIdx.x * 8 + wid;          // 0 .. 511
    int b    = it >> 7;                       // / (N/16)
    int i0   = (it & 127) * 16;
    int rb   = b * NN;

    float accL[8] = {0,0,0,0,0,0,0,0};
    float accG[8] = {0,0,0,0,0,0,0,0};
    int Mh = (lane & 16) ? 8 : 0;
    int Nl = lane & 15;

    for (int jt = 0; jt < NN / 16; ++jt) {
        int j0 = jt * 16;
        v8f cl = {}, cg = {};
#pragma unroll 4
        for (int k0 = 0; k0 < DD; k0 += 32) {
            cl = wmma_bf16(load_a_bf16(z_l, DD, rb + i0, k0, lane),
                           load_b_bf16(z_l, DD, k0, rb + j0, lane), cl);
            cg = wmma_bf16(load_a_bf16(z_g, DD, rb + i0, k0, lane),
                           load_b_bf16(z_g, DD, k0, rb + j0, lane), cg);
        }
#pragma unroll
        for (int r = 0; r < 8; ++r) {
            int gi = i0 + Mh + r, gj = j0 + Nl;
            // cosine adjacency
            float a = fmaxf(cl[r] * invn_l[rb + gi] * invn_l[rb + gj], 0.f);
            // rbf adjacency (sigma = 1)
            float d2 = fmaxf(sq_g[rb + gi] + sq_g[rb + gj] - 2.0f * cg[r], 0.f);
            float ag = __expf(-0.5f * d2);
            if (gi == gj) { a = 0.f; ag = 0.f; }
            accL[r] += a;
            accG[r] += ag;
        }
    }
    // reduce over the 16 lanes of each half (rows 0..7 live in lanes 0..15,
    // rows 8..15 in lanes 16..31; xor of bits 0..3 stays within a half)
#pragma unroll
    for (int m = 1; m <= 8; m <<= 1)
#pragma unroll
        for (int r = 0; r < 8; ++r) {
            accL[r] += __shfl_xor(accL[r], m, 32);
            accG[r] += __shfl_xor(accG[r], m, 32);
        }
    if ((lane & 15) == 0) {
#pragma unroll
        for (int r = 0; r < 8; ++r) {
            int gr = rb + i0 + Mh + r;
            invs_l[gr] = rsqrtf(fmaxf(accL[r], EPSDEG));
            invs_g[gr] = rsqrtf(fmaxf(accG[r], EPSDEG));
        }
    }
}

// ---------------------------------------------------------------------------
// Kernel 4: fused normalized-adjacency apply.
// Block = 8 waves handles one (b, 16-row i-tile) across all 512 features.
// Per 256-wide j-chunk: each wave recomputes one 16x16 Gram tile (K=512),
// applies the adjacency transform + D^{-1/2} scaling, stores the bf16 strip
// to LDS; after the barrier all waves consume the strip as the WMMA
// A-operand against zT B-fragments. Epilogue writes eta*(z - Anorm z).
// mode 0: cosine (stat = invn), mode 1: rbf (stat = sqnorm). grid = 512.
// ---------------------------------------------------------------------------
__global__ __launch_bounds__(256) void apply_kernel(
    const __bf16* __restrict__ z, const __bf16* __restrict__ zT,
    const float* __restrict__ stat, const float* __restrict__ invs,
    int mode, float eta, __bf16* __restrict__ dzs) {
    constexpr int LDA = 256 + 8;              // padded LDS stride (bf16)
    __shared__ __bf16 Alds[16 * LDA];

    int wid  = threadIdx.x >> 5;
    int lane = threadIdx.x & 31;
    int b    = blockIdx.x >> 7;
    int i0   = (blockIdx.x & 127) * 16;
    int rb   = b * NN;
    const __bf16* zTb = zT + (size_t)b * DD * NN;

    int Mh = (lane & 16) ? 8 : 0;
    int Nl = lane & 15;
    int e0w = wid * 64;                       // this wave's 64 feature cols

    v8f c0 = {}, c1 = {}, c2 = {}, c3 = {};

    for (int jc = 0; jc < NN / 256; ++jc) {
        // ---- phase 1: this wave's 16x16 Gram tile at j = jc*256 + wid*16
        int j0 = jc * 256 + wid * 16;
        v8f g = {};
#pragma unroll 4
        for (int k0 = 0; k0 < DD; k0 += 32)
            g = wmma_bf16(load_a_bf16(z, DD, rb + i0, k0, lane),
                          load_b_bf16(z, DD, k0, rb + j0, lane), g);
#pragma unroll
        for (int r = 0; r < 8; ++r) {
            int gi = i0 + Mh + r, gj = j0 + Nl;
            float a;
            if (mode == 0) {
                a = fmaxf(g[r] * stat[rb + gi] * stat[rb + gj], 0.f);
            } else {
                float d2 = fmaxf(stat[rb + gi] + stat[rb + gj] - 2.0f * g[r], 0.f);
                a = __expf(-0.5f * d2);
            }
            if (gi == gj) a = 0.f;
            a *= invs[rb + gi] * invs[rb + gj];
            Alds[(Mh + r) * LDA + wid * 16 + Nl] = (__bf16)a;
        }
        __syncthreads();

        // ---- phase 2: strip GEMM, K = 256 local j, all waves consume LDS
        int jbase = jc * 256;
#pragma unroll
        for (int kk = 0; kk < 8; ++kk) {
            v16bf a = load_a_bf16(&Alds[0], LDA, 0, kk * 32, lane);
            c0 = wmma_bf16(a, load_b_bf16(zTb, NN, jbase + kk * 32, e0w +  0, lane), c0);
            c1 = wmma_bf16(a, load_b_bf16(zTb, NN, jbase + kk * 32, e0w + 16, lane), c1);
            c2 = wmma_bf16(a, load_b_bf16(zTb, NN, jbase + kk * 32, e0w + 32, lane), c2);
            c3 = wmma_bf16(a, load_b_bf16(zTb, NN, jbase + kk * 32, e0w + 48, lane), c3);
        }
        __syncthreads();
    }

    // ---- epilogue: dzs = eta * (z - s)
    v8f cs[4] = {c0, c1, c2, c3};
#pragma unroll
    for (int t = 0; t < 4; ++t) {
#pragma unroll
        for (int r = 0; r < 8; ++r) {
            int gi = i0 + Mh + r;
            int e  = e0w + t * 16 + Nl;
            float zv = (float)z[(size_t)(rb + gi) * DD + e];
            dzs[(size_t)(rb + gi) * DD + e] = (__bf16)(eta * (zv - cs[t][r]));
        }
    }
}

// ---------------------------------------------------------------------------
// Kernel 5: out = h - dzs_l @ W_ol^T - dzs_g @ W_og^T  (eta folded into dzs).
// One 16x16 out tile per wave, chained C accumulation across both K loops.
// grid = 2048.
// ---------------------------------------------------------------------------
__global__ __launch_bounds__(256) void out_kernel(
    const float* __restrict__ h,
    const float* __restrict__ W_ol, const float* __restrict__ W_og,
    const __bf16* __restrict__ dzs_l, const __bf16* __restrict__ dzs_g,
    float* __restrict__ out) {
    int wid  = threadIdx.x >> 5;
    int lane = threadIdx.x & 31;
    int tile = blockIdx.x * 8 + wid;
    int m0   = (tile >> 5) * 16;
    int d0   = (tile & 31) * 16;

    v8f c = {};
#pragma unroll 4
    for (int k0 = 0; k0 < DD; k0 += 32)
        c = wmma_bf16(load_a_bf16(dzs_l, DD, m0, k0, lane),
                      load_b_f32(W_ol, DD, k0, d0, lane), c);
#pragma unroll 4
    for (int k0 = 0; k0 < DD; k0 += 32)
        c = wmma_bf16(load_a_bf16(dzs_g, DD, m0, k0, lane),
                      load_b_f32(W_og, DD, k0, d0, lane), c);

    int rbase = m0 + ((lane & 16) ? 8 : 0);
    int col   = d0 + (lane & 15);
#pragma unroll
    for (int r = 0; r < 8; ++r) {
        size_t idx = (size_t)(rbase + r) * DD + col;
        out[idx] = h[idx] - c[r];
    }
}

// ---------------------------------------------------------------------------
extern "C" void kernel_launch(void* const* d_in, const int* in_sizes, int n_in,
                              void* d_out, int out_size, void* d_ws, size_t ws_size,
                              hipStream_t stream) {
    const float* h    = (const float*)d_in[0];
    const float* W_pl = (const float*)d_in[1];
    const float* W_pg = (const float*)d_in[2];
    const float* W_ol = (const float*)d_in[3];
    const float* W_og = (const float*)d_in[4];
    float*       out  = (float*)d_out;

    const size_t ZB = (size_t)BATCH * NN * DD * sizeof(__bf16);   // 8,388,608 B
    const size_t SB = (size_t)BATCH * NN * sizeof(float);         // 32,768 B
    unsigned char* w = (unsigned char*)d_ws;
    __bf16* z_l    = (__bf16*)(w + 0 * ZB);
    __bf16* zT_l   = (__bf16*)(w + 1 * ZB);
    __bf16* z_g    = (__bf16*)(w + 2 * ZB);
    __bf16* zT_g   = (__bf16*)(w + 3 * ZB);
    __bf16* dzs_l  = (__bf16*)(w + 4 * ZB);
    __bf16* dzs_g  = (__bf16*)(w + 5 * ZB);
    float*  invn_l = (float*)(w + 6 * ZB);
    float*  sq_g   = (float*)(w + 6 * ZB + 1 * SB);
    float*  invs_l = (float*)(w + 6 * ZB + 2 * SB);
    float*  invs_g = (float*)(w + 6 * ZB + 3 * SB);
    // total workspace: 6*ZB + 4*SB ~= 50.5 MB

    const float eta_l = 1.0f - TWAKE;   // 0.9 (mode='wake')
    const float eta_g = TWAKE;          // 0.1

    // 1) projections
    proj_kernel<<<2048, 256, 0, stream>>>(h, W_pl, z_l, zT_l);
    proj_kernel<<<2048, 256, 0, stream>>>(h, W_pg, z_g, zT_g);
    // 2) per-row stats
    stats_kernel<<<(BATCH * NN + 255) / 256, 256, 0, stream>>>(z_l, z_g, invn_l, sq_g);
    // 3) degrees -> D^{-1/2}
    degree_kernel<<<64, 256, 0, stream>>>(z_l, z_g, invn_l, sq_g, invs_l, invs_g);
    // 4) fused normalized-adjacency apply -> eta * (z - Anorm z)
    apply_kernel<<<BATCH * (NN / 16), 256, 0, stream>>>(z_l, zT_l, invn_l, invs_l, 0, eta_l, dzs_l);
    apply_kernel<<<BATCH * (NN / 16), 256, 0, stream>>>(z_g, zT_g, sq_g,   invs_g, 1, eta_g, dzs_g);
    // 5) output projection + residual
    out_kernel<<<2048, 256, 0, stream>>>(h, W_ol, W_og, dzs_l, dzs_g, out);
}
